// DbrxAttention_35957466202271
// MI455X (gfx1250) — compile-verified, matrix operand order checked by reading, not run on hardware
//
#include <hip/hip_runtime.h>
#include <hip/hip_bf16.h>

// ---------------------------------------------------------------------------
// DBRX attention block on gfx1250 (MI455X): bf16 WMMA + TDM/async-LDS staging.
// ---------------------------------------------------------------------------

typedef __attribute__((ext_vector_type(16))) __bf16        v16bf;
typedef __attribute__((ext_vector_type(2)))  __bf16        v2bf;
typedef __attribute__((ext_vector_type(8)))  float         v8f;
typedef __attribute__((ext_vector_type(4)))  unsigned int  u32x4;
typedef __attribute__((ext_vector_type(4)))  int           i32x4;
typedef __attribute__((ext_vector_type(8)))  int           i32x8;
typedef __attribute__((ext_vector_type(4)))  float         f32x4;

#define T_SEQ    2048
#define DMODEL   6144
#define NHEADS   48
#define NKV      8
#define HDIM     128
#define QKV_N    ((NHEADS + 2 * NKV) * HDIM)   // 8192
#define CLIP_V   8.0f

// ---- CDNA5 feature gates (target-aware __has_builtin; host pass -> false) --
#if defined(__HIP_DEVICE_COMPILE__) && defined(__gfx1250__)
#if __has_builtin(__builtin_amdgcn_tensor_load_to_lds) && \
    __has_builtin(__builtin_amdgcn_s_wait_tensorcnt)
#define USE_TDM 1
#if (__clang_major__ <= 22)
#define USE_TDM_5ARG 1         // ROCm 7.2 / clang-22 form
#else
#define USE_TDM_6ARG 1         // amdgpu-toolchain / clang-23 form
#endif
#endif
#if __has_builtin(__builtin_amdgcn_global_load_async_to_lds_b128) && \
    __has_builtin(__builtin_amdgcn_s_wait_asynccnt)
#define USE_ASYNC_LDS 1
#endif
#if __has_builtin(__builtin_amdgcn_cvt_pk_bf16_f32)
#define USE_CVT_PK 1
#endif
#endif

#ifdef USE_ASYNC_LDS
typedef __attribute__((address_space(1))) i32x4 as1_i32x4;   // global int4
typedef __attribute__((address_space(3))) i32x4 as3_i32x4;   // LDS int4
#endif

union FragB16 {
    v16bf          bf;
    u32x4          q[2];
    unsigned short us[16];
};

__device__ __forceinline__ unsigned short f32_to_bf16(float f) {
    unsigned int u = __builtin_bit_cast(unsigned int, f);
    u += 0x7FFFu + ((u >> 16) & 1u);          // round-to-nearest-even
    return (unsigned short)(u >> 16);
}

__device__ __forceinline__ unsigned int cvt2_bf16(float a, float b) {
#ifdef USE_CVT_PK
    v2bf r = __builtin_amdgcn_cvt_pk_bf16_f32(a, b);
    return __builtin_bit_cast(unsigned int, r);
#else
    return (unsigned int)f32_to_bf16(a) | ((unsigned int)f32_to_bf16(b) << 16);
#endif
}

// ---------------------------------------------------------------------------
// GEMM: C[M,N] = A[M,K] * B[N,K]^T   (B is a row-major weight matrix)
// A is f32 (converted while staging) or pre-converted bf16; B is f32.
// Block tile 128x128, BK=32, 256 threads (8 waves), wave tile 32x64.
// ---------------------------------------------------------------------------
template <bool A_IS_BF16, bool DO_CLIP>
__global__ __launch_bounds__(256)
void gemm_bf16_wmma(const void* __restrict__ Ap, const float* __restrict__ B,
                    float* __restrict__ C, int M, int N, int K) {
    constexpr int BM = 128, BN = 128, BK = 32;
    constexpr int LDT = BK + 8;                    // 40 halves/row -> bank-safe
    __shared__ unsigned short a_sm[BM * LDT];
    __shared__ unsigned short b_sm[BN * LDT];
    (void)M;

    const int tid  = threadIdx.x;
    const int lane = tid & 31;
    const int wave = tid >> 5;
    const int hi   = lane >> 4, ln = lane & 15;

    const int m_blk = blockIdx.y * BM;
    const int n_blk = blockIdx.x * BN;
    const int wm    = (wave & 3) * 32;             // 4 waves along M
    const int wn    = (wave >> 2) * 64;            // 2 waves along N

    v8f acc[2][4];
#pragma unroll
    for (int i = 0; i < 2; ++i)
#pragma unroll
        for (int j = 0; j < 4; ++j)
#pragma unroll
            for (int e = 0; e < 8; ++e) acc[i][j][e] = 0.0f;

    const int srow = tid >> 1;                     // 0..127
    const int scol = (tid & 1) * 16;               // 0 or 16

    for (int k0 = 0; k0 < K; k0 += BK) {
        __syncthreads();
        // ---- stage A tile ----
        if (A_IS_BF16) {
            const unsigned short* g = (const unsigned short*)Ap +
                                      (size_t)(m_blk + srow) * K + k0 + scol;
#ifdef USE_ASYNC_LDS
            __builtin_amdgcn_global_load_async_to_lds_b128(
                (as1_i32x4*)g, (as3_i32x4*)&a_sm[srow * LDT + scol], 0, 0);
            __builtin_amdgcn_global_load_async_to_lds_b128(
                (as1_i32x4*)(g + 8), (as3_i32x4*)&a_sm[srow * LDT + scol + 8], 0, 0);
#else
            unsigned short* dst = &a_sm[srow * LDT + scol];
            const u32x4* src = (const u32x4*)g;
            ((u32x4*)dst)[0] = src[0];
            ((u32x4*)dst)[1] = src[1];
#endif
        } else {
            const f32x4* src = (const f32x4*)((const float*)Ap +
                               (size_t)(m_blk + srow) * K + k0 + scol);
            f32x4 r0 = src[0], r1 = src[1], r2 = src[2], r3 = src[3];
            u32x4 lo = {cvt2_bf16(r0[0], r0[1]), cvt2_bf16(r0[2], r0[3]),
                        cvt2_bf16(r1[0], r1[1]), cvt2_bf16(r1[2], r1[3])};
            u32x4 hi4 = {cvt2_bf16(r2[0], r2[1]), cvt2_bf16(r2[2], r2[3]),
                         cvt2_bf16(r3[0], r3[1]), cvt2_bf16(r3[2], r3[3])};
            unsigned short* dst = &a_sm[srow * LDT + scol];
            ((u32x4*)dst)[0] = lo;
            ((u32x4*)dst)[1] = hi4;
        }
        // ---- stage B tile (always f32 weights -> bf16) ----
        {
            const f32x4* src = (const f32x4*)(B + (size_t)(n_blk + srow) * K + k0 + scol);
            f32x4 r0 = src[0], r1 = src[1], r2 = src[2], r3 = src[3];
            if (k0 + BK < K)                        // global_prefetch_b8
                __builtin_prefetch(B + (size_t)(n_blk + srow) * K + k0 + BK + scol, 0, 1);
            u32x4 lo = {cvt2_bf16(r0[0], r0[1]), cvt2_bf16(r0[2], r0[3]),
                        cvt2_bf16(r1[0], r1[1]), cvt2_bf16(r1[2], r1[3])};
            u32x4 hi4 = {cvt2_bf16(r2[0], r2[1]), cvt2_bf16(r2[2], r2[3]),
                         cvt2_bf16(r3[0], r3[1]), cvt2_bf16(r3[2], r3[3])};
            unsigned short* dst = &b_sm[srow * LDT + scol];
            ((u32x4*)dst)[0] = lo;
            ((u32x4*)dst)[1] = hi4;
        }
#if defined(USE_ASYNC_LDS)
        if (A_IS_BF16) __builtin_amdgcn_s_wait_asynccnt(0);
#endif
        __syncthreads();

        // ---- fragments + 8 WMMAs ----
        const int koff = hi * 8;                   // per ISA A/B lane-half K split
        FragB16 afr[2], bfr[4];
#pragma unroll
        for (int im = 0; im < 2; ++im) {
            const unsigned short* p = &a_sm[(wm + im * 16 + ln) * LDT + koff];
            afr[im].q[0] = *(const u32x4*)p;
            afr[im].q[1] = *(const u32x4*)(p + 16);
        }
#pragma unroll
        for (int in = 0; in < 4; ++in) {
            const unsigned short* p = &b_sm[(wn + in * 16 + ln) * LDT + koff];
            bfr[in].q[0] = *(const u32x4*)p;
            bfr[in].q[1] = *(const u32x4*)(p + 16);
        }
#pragma unroll
        for (int im = 0; im < 2; ++im)
#pragma unroll
            for (int in = 0; in < 4; ++in)
                acc[im][in] = __builtin_amdgcn_wmma_f32_16x16x32_bf16(
                    false, afr[im].bf, false, bfr[in].bf,
                    (short)0, acc[im][in], false, false);
    }

    // ---- epilogue (optional clip fused) ----
#pragma unroll
    for (int im = 0; im < 2; ++im)
#pragma unroll
        for (int in = 0; in < 4; ++in)
#pragma unroll
            for (int i = 0; i < 8; ++i) {
                int row = m_blk + wm + im * 16 + i + hi * 8;
                int col = n_blk + wn + in * 16 + ln;
                float v = acc[im][in][i];
                if (DO_CLIP) v = fminf(fmaxf(v, -CLIP_V), CLIP_V);
                C[(size_t)row * N + col] = v;
            }
}

// ---------------------------------------------------------------------------
// RoPE (NeoX) + split into head-major bf16 Q/K/V.
// ---------------------------------------------------------------------------
__global__ __launch_bounds__(64)
void rope_split_kernel(const float* __restrict__ qkv,
                       const long long* __restrict__ pos_ids,
                       unsigned short* __restrict__ Qbf,
                       unsigned short* __restrict__ Kbf,
                       unsigned short* __restrict__ Vbf) {
    const int t  = blockIdx.y;
    const int hh = blockIdx.x;          // 0..47 q heads, 48..55 k, 56..63 v
    const int d  = threadIdx.x;         // 0..63
    const float p   = (float)pos_ids[t];
    const float inv = __powf(500000.0f, -(float)d * (1.0f / 64.0f));
    float sn, cs;
    __sincosf(p * inv, &sn, &cs);

    const float* src = qkv + (size_t)t * QKV_N + hh * HDIM;
    if (hh < NHEADS) {
        float x1 = src[d], x2 = src[d + 64];
        unsigned short* dst = Qbf + ((size_t)hh * T_SEQ + t) * HDIM;
        dst[d]      = f32_to_bf16(x1 * cs - x2 * sn);
        dst[d + 64] = f32_to_bf16(x2 * cs + x1 * sn);
    } else if (hh < NHEADS + NKV) {
        int kvh = hh - NHEADS;
        float x1 = src[d], x2 = src[d + 64];
        unsigned short* dst = Kbf + ((size_t)kvh * T_SEQ + t) * HDIM;
        dst[d]      = f32_to_bf16(x1 * cs - x2 * sn);
        dst[d + 64] = f32_to_bf16(x2 * cs + x1 * sn);
    } else {
        int kvh = hh - NHEADS - NKV;
        unsigned short* dst = Vbf + ((size_t)kvh * T_SEQ + t) * HDIM;
        dst[d]      = f32_to_bf16(src[d]);
        dst[d + 64] = f32_to_bf16(src[d + 64]);
    }
}

// ---------------------------------------------------------------------------
// Causal flash attention. block = (head, 128 q rows), 8 waves x 16-row q-tile.
// K chunk staged via TDM (D# with pad fields matching the LDS row padding),
// falling back to async-to-LDS, falling back to plain copies.
// ---------------------------------------------------------------------------
__global__ __launch_bounds__(256)
void attn_fa_wmma(const unsigned short* __restrict__ Qbf,
                  const unsigned short* __restrict__ Kbf,
                  const unsigned short* __restrict__ Vbf,
                  unsigned short* __restrict__ Obf) {
    constexpr int LKB = HDIM + 8;       // 136 halves/key row (TDM pad: 4 dwords)
    constexpr int LVB = 32 + 8;         // 40 halves per d row (V transposed)
    constexpr int LPB = 40;             // per-wave P staging row pitch
    __shared__ unsigned short kbuf[32 * LKB];
    __shared__ unsigned short vbuf[HDIM * LVB];
    __shared__ unsigned short pbuf[8 * 16 * LPB];

    const int tid  = threadIdx.x;
    const int lane = tid & 31;
    const int wave = tid >> 5;
    const int hi   = lane >> 4, ln = lane & 15;

    const int h    = blockIdx.y;
    const int kvh  = h / (NHEADS / NKV);
    const int qblk = blockIdx.x * 128;
    const int qt   = qblk + wave * 16;
    const float scale = 0.08838834764831845f;   // 1/sqrt(128)

#ifdef USE_TDM
    const unsigned kbuf_lds = (unsigned)(size_t)(const void*)&kbuf[0];
#endif

    // Q fragments for this wave's 16 rows (d = 0..127 -> 4 k-steps of 32)
    FragB16 qfr[4];
    {
        const unsigned short* qb = Qbf + ((size_t)h * T_SEQ + (qt + ln)) * HDIM;
#pragma unroll
        for (int kk = 0; kk < 4; ++kk) {
            int koff = kk * 32 + hi * 8;
            qfr[kk].q[0] = *(const u32x4*)(qb + koff);
            qfr[kk].q[1] = *(const u32x4*)(qb + koff + 16);
        }
    }

    v8f oacc[8];
#pragma unroll
    for (int f = 0; f < 8; ++f)
#pragma unroll
        for (int e = 0; e < 8; ++e) oacc[f][e] = 0.0f;
    float mrun[8], lrun[8];
#pragma unroll
    for (int i = 0; i < 8; ++i) { mrun[i] = -1e30f; lrun[i] = 0.0f; }

    const int kend = qblk + 128;                 // block-uniform causal bound
    const int skey = tid >> 3;                   // 0..31
    const int sseg = (tid & 7) * 16;             // 0..112

    for (int kc = 0; kc < kend; kc += 32) {
        __syncthreads();
        // ---- stage K chunk: kbuf[key][d], rows padded by 16B ----
#if defined(USE_TDM)
        if (wave == 0) {
            unsigned long long ga = (unsigned long long)(const void*)
                (Kbf + ((size_t)kvh * T_SEQ + kc) * HDIM);
            u32x4 g0;
            g0[0] = 1u;                                           // count=1, user mode
            g0[1] = kbuf_lds;                                     // lds_addr (bytes)
            g0[2] = (unsigned)(ga & 0xFFFFFFFFull);               // global_addr lo
            g0[3] = (unsigned)((ga >> 32) & 0x01FFFFFFull) | (2u << 30); // hi | type=2
            i32x8 g1;
            g1[0] = (int)((1u << 16) |        // data_size = 2 bytes
                          (1u << 20) |        // pad_enable
                          (5u << 22) |        // pad_interval: 64 dwords (one row)
                          (3u << 25));        // pad_amount: 4 dwords (16 bytes)
            g1[1] = (int)(128u << 16);        // tensor_dim0 = 128 -> bits [79:48]
            g1[2] = (int)(32u << 16);         // tensor_dim1 = 32  -> bits [111:80]
            g1[3] = (int)(128u << 16);        // tile_dim0 = 128   -> bits [127:112]
            g1[4] = 32;                       // tile_dim1 = 32
            g1[5] = 128;                      // tensor_dim0_stride = 128 elements
            g1[6] = 0;
            g1[7] = 0;
            i32x4 z4 = {0, 0, 0, 0};
#if defined(USE_TDM_5ARG)
            __builtin_amdgcn_tensor_load_to_lds(g0, g1, z4, z4, 0);
#else
            i32x8 z8 = {0, 0, 0, 0, 0, 0, 0, 0};
            __builtin_amdgcn_tensor_load_to_lds(g0, g1, z4, z4, z8, 0);
#endif
        }
#elif defined(USE_ASYNC_LDS)
        {
            const unsigned short* g = Kbf + ((size_t)kvh * T_SEQ + kc + skey) * HDIM + sseg;
            __builtin_amdgcn_global_load_async_to_lds_b128(
                (as1_i32x4*)g, (as3_i32x4*)&kbuf[skey * LKB + sseg], 0, 0);
            __builtin_amdgcn_global_load_async_to_lds_b128(
                (as1_i32x4*)(g + 8), (as3_i32x4*)&kbuf[skey * LKB + sseg + 8], 0, 0);
        }
#else
        {
            const u32x4* src = (const u32x4*)(Kbf +
                ((size_t)kvh * T_SEQ + kc + skey) * HDIM + sseg);
            u32x4* dst = (u32x4*)&kbuf[skey * LKB + sseg];
            dst[0] = src[0];
            dst[1] = src[1];
        }
#endif
        {   // stage V chunk transposed: vbuf[d][key]
            const u32x4* src = (const u32x4*)(Vbf +
                ((size_t)kvh * T_SEQ + kc + skey) * HDIM + sseg);
            u32x4 v0 = src[0], v1 = src[1];
            unsigned short tmp[16];
            *(u32x4*)&tmp[0] = v0;
            *(u32x4*)&tmp[8] = v1;
#pragma unroll
            for (int j = 0; j < 16; ++j)
                vbuf[(sseg + j) * LVB + skey] = tmp[j];
        }
#if defined(USE_TDM)
        __builtin_amdgcn_s_wait_tensorcnt(0);
#elif defined(USE_ASYNC_LDS)
        __builtin_amdgcn_s_wait_asynccnt(0);
#endif
        __syncthreads();

        // ---- S = Q K^T (two 16x16 score tiles, 4 WMMA each) ----
        float s[2][8];
#pragma unroll
        for (int n = 0; n < 2; ++n) {
            v8f sa;
#pragma unroll
            for (int e = 0; e < 8; ++e) sa[e] = 0.0f;
#pragma unroll
            for (int kk = 0; kk < 4; ++kk) {
                FragB16 kf;
                const unsigned short* p = &kbuf[(n * 16 + ln) * LKB + kk * 32 + hi * 8];
                kf.q[0] = *(const u32x4*)p;
                kf.q[1] = *(const u32x4*)(p + 16);
                sa = __builtin_amdgcn_wmma_f32_16x16x32_bf16(
                    false, qfr[kk].bf, false, kf.bf, (short)0, sa, false, false);
            }
            const int keycol = kc + n * 16 + ln;
#pragma unroll
            for (int i = 0; i < 8; ++i) {
                int qrow = qt + i + hi * 8;
                s[n][i] = (keycol <= qrow) ? sa[i] * scale : -1e30f;
            }
        }

        // ---- online softmax (row reductions across 16-lane half-groups) ----
        float mnew[8];
#pragma unroll
        for (int i = 0; i < 8; ++i) {
            float mc = fmaxf(s[0][i], s[1][i]);
            mc = fmaxf(mc, __shfl_xor(mc, 1));
            mc = fmaxf(mc, __shfl_xor(mc, 2));
            mc = fmaxf(mc, __shfl_xor(mc, 4));
            mc = fmaxf(mc, __shfl_xor(mc, 8));
            mnew[i] = fmaxf(mrun[i], mc);
        }
        unsigned short* pw = &pbuf[wave * 16 * LPB];
        float rsum[8];
#pragma unroll
        for (int i = 0; i < 8; ++i) {
            float p0 = __expf(s[0][i] - mnew[i]);
            float p1 = __expf(s[1][i] - mnew[i]);
            rsum[i]  = p0 + p1;
            pw[(i + hi * 8) * LPB + ln]      = f32_to_bf16(p0);
            pw[(i + hi * 8) * LPB + 16 + ln] = f32_to_bf16(p1);
            float alpha = __expf(mrun[i] - mnew[i]);
            lrun[i] *= alpha;
            mrun[i]  = mnew[i];
#pragma unroll
            for (int f = 0; f < 8; ++f) oacc[f][i] *= alpha;
        }
#pragma unroll
        for (int i = 0; i < 8; ++i) {
            float r = rsum[i];
            r += __shfl_xor(r, 1);
            r += __shfl_xor(r, 2);
            r += __shfl_xor(r, 4);
            r += __shfl_xor(r, 8);
            lrun[i] += r;
        }

        // ---- O += P V (P as A-fragment from per-wave LDS; DS in-order) ----
        FragB16 pf;
        {
            const unsigned short* p = &pw[ln * LPB + hi * 8];
            pf.q[0] = *(const u32x4*)p;
            pf.q[1] = *(const u32x4*)(p + 16);
        }
#pragma unroll
        for (int f = 0; f < 8; ++f) {
            FragB16 vf;
            const unsigned short* p = &vbuf[(f * 16 + ln) * LVB + hi * 8];
            vf.q[0] = *(const u32x4*)p;
            vf.q[1] = *(const u32x4*)(p + 16);
            oacc[f] = __builtin_amdgcn_wmma_f32_16x16x32_bf16(
                false, pf.bf, false, vf.bf, (short)0, oacc[f], false, false);
        }
    }

    // ---- normalize + write bf16 attn output in [T, DMODEL] layout ----
#pragma unroll
    for (int f = 0; f < 8; ++f)
#pragma unroll
        for (int i = 0; i < 8; ++i) {
            int row = qt + i + hi * 8;
            float v = oacc[f][i] / lrun[i];
            Obf[(size_t)row * DMODEL + h * HDIM + f * 16 + ln] = f32_to_bf16(v);
        }
}

// ---------------------------------------------------------------------------
extern "C" void kernel_launch(void* const* d_in, const int* in_sizes, int n_in,
                              void* d_out, int out_size, void* d_ws, size_t ws_size,
                              hipStream_t stream) {
    (void)in_sizes; (void)n_in; (void)out_size; (void)ws_size;
    const float*     hidden = (const float*)d_in[0];
    const long long* pos    = (const long long*)d_in[1];
    const float*     Wqkv   = (const float*)d_in[2];
    const float*     Wout   = (const float*)d_in[3];
    float*           out    = (float*)d_out;

    char* ws = (char*)d_ws;
    float* qkv_f32 = (float*)ws;               ws += (size_t)T_SEQ * QKV_N * 4;
    unsigned short* Qbf = (unsigned short*)ws; ws += (size_t)NHEADS * T_SEQ * HDIM * 2;
    unsigned short* Kbf = (unsigned short*)ws; ws += (size_t)NKV * T_SEQ * HDIM * 2;
    unsigned short* Vbf = (unsigned short*)ws; ws += (size_t)NKV * T_SEQ * HDIM * 2;
    unsigned short* Abf = (unsigned short*)ws; ws += (size_t)T_SEQ * DMODEL * 2;

    // 1) qkv = clip(hidden @ Wqkv^T, +-8)
    gemm_bf16_wmma<false, true><<<dim3(QKV_N / 128, T_SEQ / 128), 256, 0, stream>>>(
        hidden, Wqkv, qkv_f32, T_SEQ, QKV_N, DMODEL);
    // 2) RoPE + split to bf16 head-major Q/K/V
    rope_split_kernel<<<dim3(NHEADS + 2 * NKV, T_SEQ), 64, 0, stream>>>(
        qkv_f32, pos, Qbf, Kbf, Vbf);
    // 3) causal flash attention -> bf16 [T, DMODEL]
    attn_fa_wmma<<<dim3(T_SEQ / 128, NHEADS), 256, 0, stream>>>(Qbf, Kbf, Vbf, Abf);
    // 4) out = attn @ Wout^T
    gemm_bf16_wmma<true, false><<<dim3(DMODEL / 128, T_SEQ / 128), 256, 0, stream>>>(
        Abf, Wout, out, T_SEQ, DMODEL, DMODEL);
}
